// DeformVAE_78323023610519
// MI455X (gfx1250) — compile-verified, matrix operand order checked by reading, not run on hardware
//
#include <hip/hip_runtime.h>
#include <math.h>

// ---------------------------------------------------------------------------
// CDNA5 (gfx1250) wave32 WMMA + TDM types
// ---------------------------------------------------------------------------
typedef __attribute__((ext_vector_type(16))) _Float16 v16h;
typedef __attribute__((ext_vector_type(8)))  float    v8f;
typedef __attribute__((ext_vector_type(4)))  unsigned int v4u;
typedef __attribute__((ext_vector_type(8)))  unsigned int v8u;

// A/B fragment element -> K index mapping for V_WMMA_F32_16X16X32_F16
// (16-bit A-matrix 16x32 layout, ISA 7.12.2). Element e (0..15), lane l:
//   K = (e&7) + ((e>>3)<<4) + ((l&16)>>1)
__device__ __forceinline__ int akidx(int lane, int e) {
    return (e & 7) + ((e >> 3) << 4) + ((lane & 16) >> 1);
}

// activation codes: 0=none 1=relu 2=celu 3=sigmoid(prescale*x)
__device__ __forceinline__ float actf(float v, int act, float prescale) {
    if (act == 1) return v > 0.f ? v : 0.f;
    if (act == 2) return v > 0.f ? v : expm1f(v);
    if (act == 3) return 1.f / (1.f + expf(-v * prescale));
    return v;
}

// ---------------------------------------------------------------------------
// Tensor Data Mover: DMA a 2D fp32 tile (tile_d0 contiguous elems x tile_d1
// rows, row stride stride_el elements) from global into LDS at lds_off.
// D# per CDNA5 ISA ch.8; OOB reads return zero -> free edge zero-fill.
// ---------------------------------------------------------------------------
__device__ __forceinline__ void tdm_load_2d(unsigned lds_off, const void* gaddr,
                                            int tile_d0, int tile_d1,
                                            int tens_d0, int tens_d1,
                                            long stride_el) {
    if (tens_d0 < 0) tens_d0 = 0;
    if (tens_d1 < 0) tens_d1 = 0;
    unsigned long ga = (unsigned long)gaddr;
    v4u g0;
    g0[0] = 1u;                                           // count=1, user D#
    g0[1] = lds_off;                                      // lds_addr [63:32]
    g0[2] = (unsigned)ga;                                 // global_addr[31:0]
    g0[3] = (unsigned)((ga >> 32) & 0x01ffffffu) | (2u << 30); // addr[56:32] | type=2
    v8u g1;
    g1[0] = 2u << 16;                                     // data_size = 4B
    g1[1] = ((unsigned)tens_d0 & 0xffffu) << 16;          // tensor_dim0[15:0] @63:48
    g1[2] = (((unsigned)tens_d0 >> 16) & 0xffffu) |
            (((unsigned)tens_d1 & 0xffffu) << 16);        // dim0 hi | dim1 lo
    g1[3] = (((unsigned)tens_d1 >> 16) & 0xffffu) |
            ((unsigned)tile_d0 << 16);                    // dim1 hi | tile_dim0
    g1[4] = (unsigned)tile_d1 & 0xffffu;                  // tile_dim1 (tile_dim2=0)
    g1[5] = (unsigned)(stride_el & 0xffffffffl);          // dim0_stride[31:0]
    g1[6] = (unsigned)((stride_el >> 32) & 0xffffl);      // dim0_stride[47:32]
    g1[7] = 0u;
    asm volatile("tensor_load_to_lds %0, %1" :: "s"(g0), "s"(g1) : "memory");
}
__device__ __forceinline__ void lds_reads_done() {
    asm volatile("s_wait_dscnt 0x0" ::: "memory");
}
__device__ __forceinline__ unsigned lds_off_of(const void* p) {
    return (unsigned)(unsigned long)p;    // LDS aperture: addr[31:0] = LDS offset
}

// ---------------------------------------------------------------------------
// Implicit-GEMM forward conv (NCHW in, OIHW weights). One wave computes a
// 64(pixel) x 16(cout) tile: 4 accumulators share one TDM-staged B fragment.
// Double-buffered TDM: next weight chunk DMAs while current chunk computes.
// Requirements (met by every launch below): M = Bn<<hwlog is a multiple of 64;
// all spatial dims are powers of two. A loads are branchless and batched.
// ---------------------------------------------------------------------------
__global__ void conv2d_wmma(const float* __restrict__ in, const float* __restrict__ wt,
                            const float* __restrict__ bias, float* __restrict__ out,
                            int Cin, int hinlog, int winlog,
                            int Cout, int klog, int stride, int pad,
                            int hwlog, int wlog, int act) {
    __shared__ float BsF[2][512];              // [n][t]: 16 rows x 32 contiguous K
    const int lane = threadIdx.x;
    const int nt = blockIdx.y;
    const int ksz = 1 << klog, kk2m = (1 << (2 * klog)) - 1;
    const int K = Cin << (2 * klog);
    const int Hin = 1 << hinlog, Win = 1 << winlog;
    const int cstride_log = hinlog + winlog;   // Hin*Win
    const unsigned lds0 = lds_off_of(&BsF[0][0]);

    auto issue = [&](int kkc, int bufi) {
        // weights: 2D tensor x=K (contiguous), y=cout (stride K); edge zero-fill
        tdm_load_2d(lds0 + bufi * 2048, wt + (long)(nt * 16) * K + kkc,
                    32, 16, K - kkc, Cout - nt * 16, K);
    };

    v8f acc[4] = {};
    issue(0, 0);
    for (int kk = 0; kk < K; kk += 32) {
        int cur = (kk >> 5) & 1;
        lds_reads_done();
        if (kk + 32 < K) { issue(kk + 32, cur ^ 1); __builtin_amdgcn_s_wait_tensorcnt(1); }
        else             { __builtin_amdgcn_s_wait_tensorcnt(0); }
        v16h bf;
#pragma unroll
        for (int e = 0; e < 16; ++e)
            bf[e] = (_Float16)BsF[cur][(lane & 15) * 32 + akidx(lane, e)];
#pragma unroll
        for (int s = 0; s < 4; ++s) {
            int mrow = blockIdx.x * 64 + s * 16 + (lane & 15);
            int bb = mrow >> hwlog;
            int r0 = mrow & ((1 << hwlog) - 1);
            int iy0 = (r0 >> wlog) * stride - pad;
            int ix0 = (r0 & ((1 << wlog) - 1)) * stride - pad;
            int base = bb * (Cin << cstride_log);
            v16h af;
#pragma unroll
            for (int g = 0; g < 2; ++g) {       // batch 8 loads, then convert
                float fv[8]; bool okv[8];
#pragma unroll
                for (int u = 0; u < 8; ++u) {
                    int e = g * 8 + u;
                    int gk = kk + akidx(lane, e);
                    int ci = gk >> (2 * klog);
                    int rr = gk & kk2m;
                    int iy = iy0 + (rr >> klog), ix = ix0 + (rr & (ksz - 1));
                    okv[u] = ((unsigned)iy < (unsigned)Hin) &
                             ((unsigned)ix < (unsigned)Win) & (gk < K);
                    fv[u] = in[base + (ci << cstride_log) +
                               ((iy & (Hin - 1)) << winlog) + (ix & (Win - 1))];
                }
#pragma unroll
                for (int u = 0; u < 8; ++u)
                    af[g * 8 + u] = (_Float16)(okv[u] ? fv[u] : 0.f);
            }
            acc[s] = __builtin_amdgcn_wmma_f32_16x16x32_f16(false, af, false, bf,
                                                            (short)0, acc[s], false, false);
        }
    }
    int n = nt * 16 + (lane & 15);
    if (n < Cout) {
#pragma unroll
        for (int s = 0; s < 4; ++s)
#pragma unroll
            for (int r = 0; r < 8; ++r) {
                int m = blockIdx.x * 64 + s * 16 + r + ((lane & 16) >> 1);
                int bb = m >> hwlog;
                int sp = m & ((1 << hwlog) - 1);
                float v = acc[s][r] + bias[n];
                out[((bb * Cout + n) << hwlog) + sp] = actf(v, act, 1.f);
            }
    }
}

// ---------------------------------------------------------------------------
// WMMA GEMM: out[m,n] = act(A[m,:K] @ B[:K,n] + bias[n]); 64x16 tile per wave.
// B base += blockIdx.z * bzs ; out base += blockIdx.z * ozs ; strided output.
// Requirements: M==64 here, K multiple of 32, N multiple of 16 or TDM-zeroed.
// Double-buffered TDM for B; A loads contiguous 8-float runs (vectorizable).
// ---------------------------------------------------------------------------
__global__ void gemm_wmma(const float* __restrict__ A, int lda,
                          const float* __restrict__ Bm, int ldb, long bzs,
                          const float* __restrict__ bias,
                          float* __restrict__ out, long ozs, int som, int son,
                          int M, int N, int K, int act, float prescale) {
    __shared__ float BsF[2][512];              // [t][n]: 32 rows x 16 contiguous N
    const int lane = threadIdx.x;
    const int nt = blockIdx.y;
    const float* Bp = Bm + (long)blockIdx.z * bzs;
    float* op = out + (long)blockIdx.z * ozs;
    const unsigned lds0 = lds_off_of(&BsF[0][0]);

    auto issue = [&](int kkc, int bufi) {
        tdm_load_2d(lds0 + bufi * 2048, Bp + (long)kkc * ldb + nt * 16,
                    16, 32, N - nt * 16, K - kkc, ldb);
    };

    v8f acc[4] = {};
    issue(0, 0);
    for (int kk = 0; kk < K; kk += 32) {
        int cur = (kk >> 5) & 1;
        lds_reads_done();
        if (kk + 32 < K) { issue(kk + 32, cur ^ 1); __builtin_amdgcn_s_wait_tensorcnt(1); }
        else             { __builtin_amdgcn_s_wait_tensorcnt(0); }
        v16h bf;
#pragma unroll
        for (int e = 0; e < 16; ++e)
            bf[e] = (_Float16)BsF[cur][akidx(lane, e) * 16 + (lane & 15)];
#pragma unroll
        for (int s = 0; s < 4; ++s) {
            int mrow = blockIdx.x * 64 + s * 16 + (lane & 15);
            const float* ap = A + (long)mrow * lda + kk;
            float fv[16];
#pragma unroll
            for (int e = 0; e < 16; ++e) fv[e] = ap[akidx(lane, e)];
            v16h af;
#pragma unroll
            for (int e = 0; e < 16; ++e) af[e] = (_Float16)fv[e];
            acc[s] = __builtin_amdgcn_wmma_f32_16x16x32_f16(false, af, false, bf,
                                                            (short)0, acc[s], false, false);
        }
    }
    int n = nt * 16 + (lane & 15);
    if (n < N) {
#pragma unroll
        for (int s = 0; s < 4; ++s)
#pragma unroll
            for (int r = 0; r < 8; ++r) {
                int m = blockIdx.x * 64 + s * 16 + r + ((lane & 16) >> 1);
                if (m < M) {
                    float v = acc[s][r] + (bias ? bias[n] : 0.f);
                    op[(long)m * som + (long)n * son] = actf(v, act, prescale);
                }
            }
    }
}

// ---------------------------------------------------------------------------
// Transposed conv k=4 s=2 SAME (lax conv_transpose pad (2,2)):
//   out[oy,ox] = sum_{ky≡oy(2), kx≡ox(2)} in[(oy+ky-2)/2,(ox+kx-2)/2] w[ky,kx]
// Parity class (py,px)=blockIdx.z -> GEMM with K=4*Cin. Cin is a power of two
// >= 32, so each 32-wide K chunk sits inside one tap: input address varies
// only by ci (stride Hi*Wi). Weights HWIO; double-buffered TDM staging.
// M = Bn<<hwilog is a multiple of 64 for every launch.
// ---------------------------------------------------------------------------
__global__ void convT_s2_wmma(const float* __restrict__ in, const float* __restrict__ wt,
                              const float* __restrict__ bias, float* __restrict__ out,
                              int Cin, int cinlog, int Cout,
                              int hwilog, int wilog, int act, float prescale) {
    __shared__ float BsF[2][512];              // [t][n]: 32 ci-rows x 16 cout
    const int lane = threadIdx.x;
    const int nt = blockIdx.y;
    const int py = blockIdx.z >> 1, px = blockIdx.z & 1;
    const int K = Cin << 2;
    const int Hi = 1 << (hwilog - wilog), Wi = 1 << wilog;
    const unsigned lds0 = lds_off_of(&BsF[0][0]);

    auto issue = [&](int kkc, int bufi) {
        int tap = kkc >> cinlog, ci0c = kkc & (Cin - 1);
        int ky = py + 2 * (tap >> 1), kx = px + 2 * (tap & 1);
        tdm_load_2d(lds0 + bufi * 2048,
                    wt + (long)((ky * 4 + kx) * Cin + ci0c) * Cout + nt * 16,
                    16, 32, Cout - nt * 16, Cin - ci0c, Cout);
    };

    v8f acc[4] = {};
    issue(0, 0);
    for (int kk = 0; kk < K; kk += 32) {
        int cur = (kk >> 5) & 1;
        int tap = kk >> cinlog;                 // constant within the chunk
        int ci0 = kk & (Cin - 1);
        int dy = tap >> 1, dx = tap & 1;
        lds_reads_done();
        if (kk + 32 < K) { issue(kk + 32, cur ^ 1); __builtin_amdgcn_s_wait_tensorcnt(1); }
        else             { __builtin_amdgcn_s_wait_tensorcnt(0); }
        v16h bf;
#pragma unroll
        for (int e = 0; e < 16; ++e)
            bf[e] = (_Float16)BsF[cur][akidx(lane, e) * 16 + (lane & 15)];
#pragma unroll
        for (int s = 0; s < 4; ++s) {
            int mrow = blockIdx.x * 64 + s * 16 + (lane & 15);
            int bb = mrow >> hwilog;
            int rr = mrow & ((1 << hwilog) - 1);
            int iy = (rr >> wilog) + py + dy - 1;   // (oy+ky-2)/2, oy=2ys+py
            int ix = (rr & (Wi - 1)) + px + dx - 1;
            bool ok = ((unsigned)iy < (unsigned)Hi) & ((unsigned)ix < (unsigned)Wi);
            int base = ((bb * Cin + ci0) << hwilog) +
                       ((iy & (Hi - 1)) << wilog) + (ix & (Wi - 1));
            float fv[16];
#pragma unroll
            for (int e = 0; e < 16; ++e)
                fv[e] = in[base + (akidx(lane, e) << hwilog)];
            v16h af;
#pragma unroll
            for (int e = 0; e < 16; ++e)
                af[e] = (_Float16)(ok ? fv[e] : 0.f);
            acc[s] = __builtin_amdgcn_wmma_f32_16x16x32_f16(false, af, false, bf,
                                                            (short)0, acc[s], false, false);
        }
    }
    int n = nt * 16 + (lane & 15);
    if (n < Cout) {
#pragma unroll
        for (int s = 0; s < 4; ++s)
#pragma unroll
            for (int r = 0; r < 8; ++r) {
                int m = blockIdx.x * 64 + s * 16 + r + ((lane & 16) >> 1);
                int bb = m >> hwilog;
                int q = m & ((1 << hwilog) - 1);
                int oy = 2 * (q >> wilog) + py, ox = 2 * (q & (Wi - 1)) + px;
                float v = acc[s][r] + bias[n];
                out[((bb * Cout + n) << (hwilog + 2)) + (oy << (wilog + 1)) + ox] =
                    actf(v, act, prescale);
            }
    }
}

// ---------------------------------------------------------------------------
// Latent "what" math + h1c = concat(h1, z_what). idx covers 64 x 320.
// ---------------------------------------------------------------------------
__global__ void latent_what_k(const float* __restrict__ h1, const float* __restrict__ loc,
                              const float* __restrict__ sclp, const float* __restrict__ spkp,
                              const float* __restrict__ eta, const float* __restrict__ eps,
                              float* __restrict__ h1c, float* __restrict__ zwhat,
                              float* __restrict__ klel) {
    int idx = blockIdx.x * blockDim.x + threadIdx.x;
    if (idx >= 64 * 320) return;
    int b = idx / 320, j = idx % 320;
    if (j < 256) { h1c[b * 320 + j] = h1[b * 256 + j]; return; }
    int zd = j - 256;
    int p = b * 64 + zd;
    const float pl = 0.9724924563f;                 // 0.8^(1/8)
    float sc = sclp[p];
    sc = sc > 20.f ? sc : log1pf(expf(sc));         // softplus
    float mu = pl * loc[p];
    float sd = pl * sc - pl + 1.f + 1e-15f;
    float zs = mu + sd * eps[p];
    float spre = spkp[p];
    float sp = expf(-(spre > 0.f ? spre : 0.f));
    float gamma = 1.f / (1.f + expf(-50.f * (eta[p] + sp - 1.f)));
    float zw = zs * gamma;
    zwhat[p] = zw;
    h1c[b * 320 + j] = zw;
    float spc = fminf(fmaxf(sp, 1e-15f), 1.f - 1e-15f);
    float kl_slab = spc * (-logf(sd) + 0.5f * (sd * sd + mu * mu) - 0.5f);
    float kl_bern = spc * logf(spc * 10.f) + (1.f - spc) * logf((1.f - spc) / 0.9f);
    klel[b * 128 + zd] = kl_slab + kl_bern;
}

__global__ void latent_warp_k(const float* __restrict__ wloc, const float* __restrict__ wscp,
                              const float* __restrict__ eps, float* __restrict__ zwarp,
                              float* __restrict__ klel) {
    int idx = blockIdx.x * blockDim.x + threadIdx.x;
    if (idx >= 64 * 64) return;
    int b = idx >> 6, zd = idx & 63;
    float lo = wloc[idx];
    float sc = wscp[idx];
    sc = sc > 20.f ? sc : log1pf(expf(sc));
    zwarp[idx] = lo + sc * eps[idx];
    klel[b * 128 + 64 + zd] = -logf(sc + 1e-15f) + 0.5f * (sc * sc + lo * lo) - 0.5f;
}

__global__ void kl_reduce_k(const float* __restrict__ klel, float* __restrict__ klsum) {
    __shared__ float s[128];
    int b = blockIdx.x, t = threadIdx.x;
    s[t] = klel[b * 128 + t];
    __syncthreads();
    for (int o = 64; o > 0; o >>= 1) { if (t < o) s[t] += s[t + o]; __syncthreads(); }
    if (t == 0) klsum[b] = s[0];                    // BETA = 1
}

// cumsum -> warp grid coords, in place on flow (64,2,128,128)
__global__ void cumsum_k(float* __restrict__ fl) {
    int idx = blockIdx.x * blockDim.x + threadIdx.x;
    if (idx >= 2 * 64 * 128) return;
    const float inv = 1.f / 63.5f;                  // (128-1)/2
    if (idx < 64 * 128) {                            // gx: scan along W
        int b = idx >> 7, y = idx & 127;
        float* p = fl + ((b * 2 + 0) * 128 + y) * 128;
        float cs = 0.f;
        for (int x = 0; x < 128; ++x) { cs += p[x]; p[x] = (cs * 2.f - 1.f) * inv - 1.f; }
    } else {                                         // gy: scan along H
        int j = idx - 8192;
        int b = j >> 7, x = j & 127;
        float* p = fl + ((b * 2 + 1) * 128) * 128 + x;
        float cs = 0.f;
        for (int y = 0; y < 128; ++y) { cs += p[y * 128]; p[y * 128] = (cs * 2.f - 1.f) * inv - 1.f; }
    }
}

// bilinear grid sample of xd + Gaussian log-likelihood, partial sums per block
__global__ void sample_ll_k(const float* __restrict__ xd, const float* __restrict__ fl,
                            const float* __restrict__ xin, float* __restrict__ part) {
    __shared__ float s[256];
    int b = blockIdx.x;
    int pix = blockIdx.y * 256 + threadIdx.x;
    int y = pix >> 7, x = pix & 127;
    float gx = fl[((b * 2 + 0) * 128 + y) * 128 + x];
    float gy = fl[((b * 2 + 1) * 128 + y) * 128 + x];
    gx = fminf(fmaxf(gx, -1.f), 1.f);
    gy = fminf(fmaxf(gy, -1.f), 1.f);
    float ix = ((gx + 1.f) * 128.f - 1.f) * 0.5f;
    float iy = ((gy + 1.f) * 128.f - 1.f) * 0.5f;
    float x0f = floorf(ix), y0f = floorf(iy);
    int x0 = (int)x0f, y0 = (int)y0f;
    float wa = (x0f + 1.f - ix) * (y0f + 1.f - iy);
    float wb = (ix - x0f) * (y0f + 1.f - iy);
    float wc = (x0f + 1.f - ix) * (iy - y0f);
    float wd = (ix - x0f) * (iy - y0f);
    float ll = 0.f;
    for (int c = 0; c < 3; ++c) {
        const float* img = xd + (b * 3 + c) * 16384;
        float v = 0.f;
        int yy = y0, xx = x0;
        if (yy >= 0 && yy < 128 && xx >= 0 && xx < 128) v += wa * img[yy * 128 + xx];
        xx = x0 + 1;
        if (yy >= 0 && yy < 128 && xx >= 0 && xx < 128) v += wb * img[yy * 128 + xx];
        yy = y0 + 1; xx = x0;
        if (yy >= 0 && yy < 128 && xx >= 0 && xx < 128) v += wc * img[yy * 128 + xx];
        xx = x0 + 1;
        if (yy >= 0 && yy < 128 && xx >= 0 && xx < 128) v += wd * img[yy * 128 + xx];
        float d = xin[((b * 3 + c) * 128 + y) * 128 + x] - v;
        ll += -5.5555555556f * d * d + 0.2850342711f;   // -0.5/0.3^2, -log(0.3)-0.5*log(2pi)
    }
    s[threadIdx.x] = ll;
    __syncthreads();
    for (int o = 128; o > 0; o >>= 1) {
        if (threadIdx.x < o) s[threadIdx.x] += s[threadIdx.x + o];
        __syncthreads();
    }
    if (threadIdx.x == 0) part[b * 64 + blockIdx.y] = s[0];
}

__global__ void finalize_k(const float* __restrict__ part, const float* __restrict__ klsum,
                           float* __restrict__ loss) {
    __shared__ float s[64];
    int b = blockIdx.x, t = threadIdx.x;
    s[t] = part[b * 64 + t];
    __syncthreads();
    for (int o = 32; o > 0; o >>= 1) { if (t < o) s[t] += s[t + o]; __syncthreads(); }
    if (t == 0) loss[b] = klsum[b] - s[0];          // loss = kl - log_like
}

// ---------------------------------------------------------------------------
// Host orchestration
// ---------------------------------------------------------------------------
extern "C" void kernel_launch(void* const* d_in, const int* in_sizes, int n_in,
                              void* d_out, int out_size, void* d_ws, size_t ws_size,
                              hipStream_t stream) {
    (void)in_sizes; (void)n_in; (void)out_size; (void)ws_size;
    const float* x        = (const float*)d_in[0];
    const float* eta      = (const float*)d_in[1];
    const float* eps_slab = (const float*)d_in[2];
    const float* eps_warp = (const float*)d_in[3];
    const float* ew[5]; const float* eb[5];
    const float* dw[5]; const float* db[5];
    const float* ww[5]; const float* wb[5];
    for (int i = 0; i < 5; ++i) {
        ew[i] = (const float*)d_in[4  + 2 * i]; eb[i] = (const float*)d_in[5  + 2 * i];
        dw[i] = (const float*)d_in[14 + 2 * i]; db[i] = (const float*)d_in[15 + 2 * i];
        ww[i] = (const float*)d_in[24 + 2 * i]; wb[i] = (const float*)d_in[25 + 2 * i];
    }
    const float* loc1w = (const float*)d_in[34]; const float* loc1b = (const float*)d_in[35];
    const float* loc2w = (const float*)d_in[36]; const float* loc2b = (const float*)d_in[37];
    const float* scl1w = (const float*)d_in[38]; const float* scl1b = (const float*)d_in[39];
    const float* scl2w = (const float*)d_in[40]; const float* scl2b = (const float*)d_in[41];
    const float* spk1w = (const float*)d_in[42]; const float* spk1b = (const float*)d_in[43];
    const float* spk2w = (const float*)d_in[44]; const float* spk2b = (const float*)d_in[45];
    const float* wlo1w = (const float*)d_in[46]; const float* wlo1b = (const float*)d_in[47];
    const float* wlo2w = (const float*)d_in[48]; const float* wlo2b = (const float*)d_in[49];
    const float* wsc1w = (const float*)d_in[50]; const float* wsc1b = (const float*)d_in[51];
    const float* wsc2w = (const float*)d_in[52]; const float* wsc2b = (const float*)d_in[53];

    float* W = (float*)d_ws;
    float* bufA = W;                 // 64*32*64*64   (enc1 out / dec layer4 out)
    float* bufB = bufA + 8388608;    // 64*64*32*32   (enc2 / dec layer3)
    float* bufC = bufB + 4194304;    // 64*128*16*16  (enc3 / dec layer2)
    float* bufD = bufC + 2097152;    // 64*256*8*8    (enc4 / dec layer1)
    float* XD   = bufD + 1048576;    // 64*3*128*128
    float* FL   = XD   + 3145728;    // 64*2*128*128  (flow -> grid in place)
    float* h1   = FL   + 2097152;    // 64*256
    float* hid  = h1   + 16384;      // 64*256
    float* locb = hid  + 16384;      // 64*64
    float* sclb = locb + 4096;
    float* spkb = sclb + 4096;
    float* h1c  = spkb + 4096;       // 64*320
    float* wlob = h1c  + 20480;      // 64*64
    float* wscb = wlob + 4096;
    float* zwhat= wscb + 4096;
    float* zwarp= zwhat+ 4096;
    float* klel = zwarp+ 4096;       // 64*128
    float* partb= klel + 8192;       // 64*64
    float* klsum= partb+ 4096;       // 64

    dim3 wv(32);

    // ---- encoder (Cin, hinlog, winlog, Cout, klog, stride, pad, hwlog, wlog, act) ----
    conv2d_wmma<<<dim3(4096, 2), wv, 0, stream>>>(x,    ew[0], eb[0], bufA,   3, 7, 7,  32, 2, 2, 1, 12, 6, 1);
    conv2d_wmma<<<dim3(1024, 4), wv, 0, stream>>>(bufA, ew[1], eb[1], bufB,  32, 6, 6,  64, 2, 2, 1, 10, 5, 1);
    conv2d_wmma<<<dim3(256,  8), wv, 0, stream>>>(bufB, ew[2], eb[2], bufC,  64, 5, 5, 128, 2, 2, 1,  8, 4, 1);
    conv2d_wmma<<<dim3(64,  16), wv, 0, stream>>>(bufC, ew[3], eb[3], bufD, 128, 4, 4, 256, 2, 2, 1,  6, 3, 1);
    conv2d_wmma<<<dim3(1,   16), wv, 0, stream>>>(bufD, ew[4], eb[4], h1,   256, 3, 3, 256, 3, 1, 0,  0, 0, 1);

    // ---- what-MLPs (celu hidden) ----
    gemm_wmma<<<dim3(1, 16), wv, 0, stream>>>(h1, 256, loc1w, 256, 0, loc1b, hid, 0, 256, 1, 64, 256, 256, 2, 1.f);
    gemm_wmma<<<dim3(1,  4), wv, 0, stream>>>(hid, 256, loc2w,  64, 0, loc2b, locb, 0, 64, 1, 64,  64, 256, 0, 1.f);
    gemm_wmma<<<dim3(1, 16), wv, 0, stream>>>(h1, 256, scl1w, 256, 0, scl1b, hid, 0, 256, 1, 64, 256, 256, 2, 1.f);
    gemm_wmma<<<dim3(1,  4), wv, 0, stream>>>(hid, 256, scl2w,  64, 0, scl2b, sclb, 0, 64, 1, 64,  64, 256, 0, 1.f);
    gemm_wmma<<<dim3(1, 16), wv, 0, stream>>>(h1, 256, spk1w, 256, 0, spk1b, hid, 0, 256, 1, 64, 256, 256, 2, 1.f);
    gemm_wmma<<<dim3(1,  4), wv, 0, stream>>>(hid, 256, spk2w,  64, 0, spk2b, spkb, 0, 64, 1, 64,  64, 256, 0, 1.f);

    latent_what_k<<<80, 256, 0, stream>>>(h1, locb, sclb, spkb, eta, eps_slab, h1c, zwhat, klel);

    // ---- warp-MLPs (input h1c 64x320) ----
    gemm_wmma<<<dim3(1, 16), wv, 0, stream>>>(h1c, 320, wlo1w, 256, 0, wlo1b, hid, 0, 256, 1, 64, 256, 320, 2, 1.f);
    gemm_wmma<<<dim3(1,  4), wv, 0, stream>>>(hid, 256, wlo2w,  64, 0, wlo2b, wlob, 0, 64, 1, 64,  64, 256, 0, 1.f);
    gemm_wmma<<<dim3(1, 16), wv, 0, stream>>>(h1c, 320, wsc1w, 256, 0, wsc1b, hid, 0, 256, 1, 64, 256, 320, 2, 1.f);
    gemm_wmma<<<dim3(1,  4), wv, 0, stream>>>(hid, 256, wsc2w,  64, 0, wsc2b, wscb, 0, 64, 1, 64,  64, 256, 0, 1.f);

    latent_warp_k<<<16, 256, 0, stream>>>(wlob, wscb, eps_warp, zwarp, klel);
    kl_reduce_k<<<64, 128, 0, stream>>>(klel, klsum);

    // ---- decoder (z_what -> xd) ----
    // convT k8 VALID on 1x1 input: per position z the weight plane is 63-z
    // (lax conv_transpose correlates over the zero-dilated input -> reversed kernel)
    gemm_wmma<<<dim3(1, 16, 64), wv, 0, stream>>>(zwhat, 64, dw[0] + 63l * 64 * 256, 256,
                                                  -(long)(64 * 256), db[0],
                                                  bufD, 1, 256 * 64, 64, 64, 256, 64, 1, 1.f);
    convT_s2_wmma<<<dim3(64,   8, 4), wv, 0, stream>>>(bufD, dw[1], db[1], bufC, 256, 8, 128,  6, 3, 1, 1.f);
    convT_s2_wmma<<<dim3(256,  4, 4), wv, 0, stream>>>(bufC, dw[2], db[2], bufB, 128, 7,  64,  8, 4, 1, 1.f);
    convT_s2_wmma<<<dim3(1024, 2, 4), wv, 0, stream>>>(bufB, dw[3], db[3], bufA,  64, 6,  32, 10, 5, 1, 1.f);
    convT_s2_wmma<<<dim3(4096, 1, 4), wv, 0, stream>>>(bufA, dw[4], db[4], XD,    32, 5,   3, 12, 6, 3, 1.f);

    // ---- warp decoder (z_warp -> flow), buffers reused ----
    gemm_wmma<<<dim3(1, 16, 64), wv, 0, stream>>>(zwarp, 64, ww[0] + 63l * 64 * 256, 256,
                                                  -(long)(64 * 256), wb[0],
                                                  bufD, 1, 256 * 64, 64, 64, 256, 64, 1, 1.f);
    convT_s2_wmma<<<dim3(64,   8, 4), wv, 0, stream>>>(bufD, ww[1], wb[1], bufC, 256, 8, 128,  6, 3, 1, 1.f);
    convT_s2_wmma<<<dim3(256,  4, 4), wv, 0, stream>>>(bufC, ww[2], wb[2], bufB, 128, 7,  64,  8, 4, 1, 1.f);
    convT_s2_wmma<<<dim3(1024, 2, 4), wv, 0, stream>>>(bufB, ww[3], wb[3], bufA,  64, 6,  32, 10, 5, 1, 1.f);
    convT_s2_wmma<<<dim3(4096, 1, 4), wv, 0, stream>>>(bufA, ww[4], wb[4], FL,    32, 5,   2, 12, 6, 3, 1.f /*WARP_PREV*/);

    // ---- warp grid, sampling, loss ----
    cumsum_k<<<64, 256, 0, stream>>>(FL);
    sample_ll_k<<<dim3(64, 64), 256, 0, stream>>>(XD, FL, x, partb);
    finalize_k<<<64, 64, 0, stream>>>(partb, klsum, (float*)d_out);
}
